// LSTMExtended_28922309771280
// MI455X (gfx1250) — compile-verified, matrix operand order checked by reading, not run on hardware
//
#include <hip/hip_runtime.h>

// ---------------------------------------------------------------------------
// Problem constants (from the reference)
// ---------------------------------------------------------------------------
#define B_      1024
#define H_      512
#define F_      32
#define CIN_    4
#define KCONV_  30
#define STRIDE_ 6
#define L_      924
#define T_      150
#define KD_     40
#define NSTEPS_ 10
#define G4H_    2048   // 4*H

// ---------------------------------------------------------------------------
// Types / helpers
// ---------------------------------------------------------------------------
typedef __attribute__((ext_vector_type(16))) __bf16        bf16x16;
typedef __attribute__((ext_vector_type(16))) unsigned short u16x16;
typedef __attribute__((ext_vector_type(8)))  float          f32x8;

__device__ __forceinline__ unsigned short f2bf(float f) {
  unsigned int u = __builtin_bit_cast(unsigned int, f);
  u += 0x7FFFu + ((u >> 16) & 1u);          // round-to-nearest-even
  return (unsigned short)(u >> 16);
}
__device__ __forceinline__ float bf2f(unsigned short s) {
  unsigned int u = ((unsigned int)s) << 16;
  return __builtin_bit_cast(float, u);
}
__device__ __forceinline__ float sigm(float v) { return 1.0f / (1.0f + __expf(-v)); }

// Load one wave32 WMMA bf16 operand fragment (16x32 slab of a row-major
// [rows, ldk] bf16 matrix).  Per ISA 7.12.2: lane l (0..15) holds row l,
// K = {0..7, 16..23}; lane l+16 holds row l, K = {8..15, 24..31}.
// Works for both A ([M,K] row-major) and B (= W stored [N,K] row-major,
// which is the K x N operand loaded column-major).
__device__ __forceinline__ bf16x16 load_frag(const unsigned short* __restrict__ base,
                                             int row0, int ldk, int k, int lane) {
  const unsigned short* p =
      base + (size_t)(row0 + (lane & 15)) * ldk + (size_t)(k + ((lane >> 4) << 3));
  u16x16 t;
#pragma unroll
  for (int i = 0; i < 8; ++i) { t[i] = p[i]; t[8 + i] = p[16 + i]; }
  return __builtin_bit_cast(bf16x16, t);
}

// ---------------------------------------------------------------------------
// Fused LSTM step:  gates = h_prev @ Whh^T + x @ Wx^T + bias ; cell update.
// Each wave owns a 16(batch) x 16(j) patch and computes the i/f/g/o tiles
// (W rows j, j+512, j+1024, j+1536) so the nonlinearity fuses in registers.
// Grid: 256 blocks x 256 threads = 2048 waves = (1024/16) x (512/16) tiles.
// ---------------------------------------------------------------------------
__global__ __launch_bounds__(256)
void lstm_step_kernel(const unsigned short* __restrict__ h_prev, // [B_,H_] bf16
                      const unsigned short* __restrict__ x,      // [B_,Kx] bf16
                      float*          __restrict__ c,            // [B_,H_] f32 (in/out)
                      unsigned short* __restrict__ h_out,        // [B_,H_] bf16
                      const unsigned short* __restrict__ Whh,    // [G4H_,H_] bf16
                      const unsigned short* __restrict__ Wx,     // [G4H_,Kx] bf16
                      const float* __restrict__ bias,            // [G4H_] f32
                      int Kx) {
  const int lane = threadIdx.x & 31;
  const int wave = blockIdx.x * (blockDim.x >> 5) + (threadIdx.x >> 5);
  const int b0   = (wave >> 5) << 4;   // batch tile origin  (64 tiles)
  const int j0   = (wave & 31) << 4;   // hidden tile origin (32 tiles)

  // Warm L2/L0 for the 4 weight row-blocks this wave consumes.
#pragma unroll
  for (int q = 0; q < 4; ++q)
    __builtin_prefetch(Whh + (size_t)(q * H_ + j0) * H_, 0, 3);

  f32x8 acc[4] = {};

  // Recurrent part: K = H_ = 512  -> 16 wmma k-steps, 4 gate tiles each.
  for (int k = 0; k < H_; k += 32) {
    bf16x16 a = load_frag(h_prev, b0, H_, k, lane);
#pragma unroll
    for (int q = 0; q < 4; ++q) {
      bf16x16 b = load_frag(Whh, q * H_ + j0, H_, k, lane);
      acc[q] = __builtin_amdgcn_wmma_f32_16x16x32_bf16(
          false, a, false, b, (short)0, acc[q], false, false);
    }
  }
  // Input part: Kx = 32 (layer 0 / decode-in) or 512 (layer 1).
  for (int k = 0; k < Kx; k += 32) {
    bf16x16 a = load_frag(x, b0, Kx, k, lane);
#pragma unroll
    for (int q = 0; q < 4; ++q) {
      bf16x16 b = load_frag(Wx, q * H_ + j0, Kx, k, lane);
      acc[q] = __builtin_amdgcn_wmma_f32_16x16x32_bf16(
          false, a, false, b, (short)0, acc[q], false, false);
    }
  }

  // Epilogue: C/D layout -> lane holds N = j0 + (lane&15); rows M = r + 8*(lane>=16).
  const int j  = j0 + (lane & 15);
  const int bb = b0 + ((lane >> 4) << 3);
  const float bi = bias[j];
  const float bf = bias[H_ + j];
  const float bg = bias[2 * H_ + j];
  const float bo = bias[3 * H_ + j];
#pragma unroll
  for (int r = 0; r < 8; ++r) {
    const size_t idx = (size_t)(bb + r) * H_ + j;
    const float gi = sigm(acc[0][r] + bi);
    const float gf = sigm(acc[1][r] + bf);
    const float gg = tanhf(acc[2][r] + bg);
    const float go = sigm(acc[3][r] + bo);
    const float cn = gf * c[idx] + gi * gg;
    c[idx]     = cn;
    h_out[idx] = f2bf(go * tanhf(cn));
  }
}

// ---------------------------------------------------------------------------
// Encoder conv1d + ReLU -> seq bf16 [T_, B_, F_]
// ---------------------------------------------------------------------------
__global__ void conv_encoder_kernel(const float* __restrict__ x,       // [B_,CIN_,L_]
                                    const float* __restrict__ w,       // [F_,CIN_,KCONV_]
                                    const float* __restrict__ b,       // [F_]
                                    unsigned short* __restrict__ seq) {
  const int id = blockIdx.x * blockDim.x + threadIdx.x;          // T_*B_*F_ threads
  const int f  = id & (F_ - 1);
  const int bb = (id >> 5) & (B_ - 1);
  const int t  = id >> 15;
  if (t >= T_) return;
  float acc = b[f];
  const float* xr = x + (size_t)bb * CIN_ * L_ + t * STRIDE_;
  const float* wr = w + (size_t)f * CIN_ * KCONV_;
#pragma unroll
  for (int c = 0; c < CIN_; ++c) {
#pragma unroll
    for (int k = 0; k < KCONV_; ++k)
      acc += xr[c * L_ + k] * wr[c * KCONV_ + k];
  }
  acc = acc > 0.0f ? acc : 0.0f;
  seq[(size_t)t * B_ * F_ + (size_t)bb * F_ + f] = f2bf(acc);
}

// f32 -> bf16 elementwise
__global__ void cvt_bf16_kernel(const float* __restrict__ src,
                                unsigned short* __restrict__ dst, int n) {
  int i = blockIdx.x * blockDim.x + threadIdx.x;
  if (i < n) dst[i] = f2bf(src[i]);
}

// bias = a + b
__global__ void add_bias_kernel(const float* __restrict__ a, const float* __restrict__ b,
                                float* __restrict__ o, int n) {
  int i = blockIdx.x * blockDim.x + threadIdx.x;
  if (i < n) o[i] = a[i] + b[i];
}

// zero initial LSTM state (h bf16, c f32)
__global__ void zero_state_kernel(unsigned short* h0, unsigned short* h1,
                                  float* c0, float* c1) {
  int i = blockIdx.x * blockDim.x + threadIdx.x;   // B_*H_ threads
  h0[i] = 0; h1[i] = 0; c0[i] = 0.0f; c1[i] = 0.0f;
}

// y = h1 @ dec_w^T + dec_b  (-> bf16 y, optional f32 fut[:, :, s])
__global__ void proj_kernel(const unsigned short* __restrict__ h1,  // [B_,H_] bf16
                            const float* __restrict__ dec_w,        // [F_,H_]
                            const float* __restrict__ dec_b,        // [F_]
                            unsigned short* __restrict__ y,         // [B_,F_] bf16
                            float* __restrict__ fut,                // [B_,F_,NSTEPS_] or null
                            int s) {
  const int id = blockIdx.x * blockDim.x + threadIdx.x;  // B_*F_ threads
  const int f  = id & (F_ - 1);
  const int b  = id >> 5;
  float acc = dec_b[f];
  const unsigned short* hr = h1 + (size_t)b * H_;
  const float* wr = dec_w + (size_t)f * H_;
#pragma unroll 8
  for (int k = 0; k < H_; ++k) acc += bf2f(hr[k]) * wr[k];
  y[(size_t)b * F_ + f] = f2bf(acc);
  if (fut) fut[((size_t)b * F_ + f) * NSTEPS_ + s] = acc;
}

// ConvTranspose1d (stride 1, no pad, no bias): out[b,j] = sum_{s,f} fut * w[f,j-s]
__global__ void deconv_kernel(const float* __restrict__ fut,       // [B_,F_,NSTEPS_]
                              const float* __restrict__ w,         // [F_,1,KD_]
                              float* __restrict__ out) {           // [B_,1,49]
  const int id = blockIdx.x * blockDim.x + threadIdx.x;
  const int JOUT = NSTEPS_ + KD_ - 1;                              // 49
  if (id >= B_ * JOUT) return;
  const int j = id % JOUT;
  const int b = id / JOUT;
  const int s_lo = (j - (KD_ - 1)) > 0 ? (j - (KD_ - 1)) : 0;
  const int s_hi = j < (NSTEPS_ - 1) ? j : (NSTEPS_ - 1);
  float acc = 0.0f;
  for (int s = s_lo; s <= s_hi; ++s) {
    const int kk = j - s;                                          // 0..KD_-1
#pragma unroll
    for (int f = 0; f < F_; ++f)
      acc += fut[((size_t)b * F_ + f) * NSTEPS_ + s] * w[f * KD_ + kk];
  }
  out[(size_t)b * JOUT + j] = acc;
}

// ---------------------------------------------------------------------------
// Host orchestration
// ---------------------------------------------------------------------------
extern "C" void kernel_launch(void* const* d_in, const int* in_sizes, int n_in,
                              void* d_out, int out_size, void* d_ws, size_t ws_size,
                              hipStream_t stream) {
  const float* x       = (const float*)d_in[0];
  const float* conv_w  = (const float*)d_in[1];
  const float* conv_b  = (const float*)d_in[2];
  const float* Wih0    = (const float*)d_in[3];
  const float* Whh0    = (const float*)d_in[4];
  const float* bih0    = (const float*)d_in[5];
  const float* bhh0    = (const float*)d_in[6];
  const float* Wih1    = (const float*)d_in[7];
  const float* Whh1    = (const float*)d_in[8];
  const float* bih1    = (const float*)d_in[9];
  const float* bhh1    = (const float*)d_in[10];
  const float* dec_w   = (const float*)d_in[11];
  const float* dec_b   = (const float*)d_in[12];
  const float* deconvw = (const float*)d_in[13];

  // ---- carve workspace ----
  char* p = (char*)d_ws;
  auto carve = [&](size_t bytes) {
    void* r = (void*)p;
    p += (bytes + 255) & ~(size_t)255;
    return r;
  };
  unsigned short* seq   = (unsigned short*)carve((size_t)T_ * B_ * F_ * 2);
  unsigned short* Whh0b = (unsigned short*)carve((size_t)G4H_ * H_ * 2);
  unsigned short* Wih0b = (unsigned short*)carve((size_t)G4H_ * F_ * 2);
  unsigned short* Whh1b = (unsigned short*)carve((size_t)G4H_ * H_ * 2);
  unsigned short* Wih1b = (unsigned short*)carve((size_t)G4H_ * H_ * 2);
  float*          b0    = (float*)carve((size_t)G4H_ * 4);
  float*          b1    = (float*)carve((size_t)G4H_ * 4);
  unsigned short* h0[2] = {(unsigned short*)carve((size_t)B_ * H_ * 2),
                           (unsigned short*)carve((size_t)B_ * H_ * 2)};
  unsigned short* h1[2] = {(unsigned short*)carve((size_t)B_ * H_ * 2),
                           (unsigned short*)carve((size_t)B_ * H_ * 2)};
  float*          c0    = (float*)carve((size_t)B_ * H_ * 4);
  float*          c1    = (float*)carve((size_t)B_ * H_ * 4);
  unsigned short* y     = (unsigned short*)carve((size_t)B_ * F_ * 2);
  float*          fut   = (float*)carve((size_t)B_ * F_ * NSTEPS_ * 4);

  // ---- weight conversion (f32 -> bf16) + combined biases ----
  cvt_bf16_kernel<<<(G4H_ * H_ + 255) / 256, 256, 0, stream>>>(Whh0, Whh0b, G4H_ * H_);
  cvt_bf16_kernel<<<(G4H_ * F_ + 255) / 256, 256, 0, stream>>>(Wih0, Wih0b, G4H_ * F_);
  cvt_bf16_kernel<<<(G4H_ * H_ + 255) / 256, 256, 0, stream>>>(Whh1, Whh1b, G4H_ * H_);
  cvt_bf16_kernel<<<(G4H_ * H_ + 255) / 256, 256, 0, stream>>>(Wih1, Wih1b, G4H_ * H_);
  add_bias_kernel<<<(G4H_ + 255) / 256, 256, 0, stream>>>(bih0, bhh0, b0, G4H_);
  add_bias_kernel<<<(G4H_ + 255) / 256, 256, 0, stream>>>(bih1, bhh1, b1, G4H_);

  // ---- encoder conv ----
  conv_encoder_kernel<<<(T_ * B_ * F_) / 256, 256, 0, stream>>>(x, conv_w, conv_b, seq);

  // ---- init state ----
  zero_state_kernel<<<(B_ * H_) / 256, 256, 0, stream>>>(h0[0], h1[0], c0, c1);

  // ---- encoder LSTM: 150 steps x 2 layers, stream order = recurrence sync ----
  int p0 = 0, p1 = 0;
  for (int t = 0; t < T_; ++t) {
    lstm_step_kernel<<<256, 256, 0, stream>>>(h0[p0], seq + (size_t)t * B_ * F_, c0,
                                              h0[p0 ^ 1], Whh0b, Wih0b, b0, F_);
    p0 ^= 1;
    lstm_step_kernel<<<256, 256, 0, stream>>>(h1[p1], h0[p0], c1,
                                              h1[p1 ^ 1], Whh1b, Wih1b, b1, H_);
    p1 ^= 1;
  }

  // y0 = last decoded encoder step
  proj_kernel<<<(B_ * F_) / 256, 256, 0, stream>>>(h1[p1], dec_w, dec_b, y,
                                                   (float*)nullptr, 0);

  // ---- autoregressive decode: 10 steps ----
  for (int s = 0; s < NSTEPS_; ++s) {
    lstm_step_kernel<<<256, 256, 0, stream>>>(h0[p0], y, c0,
                                              h0[p0 ^ 1], Whh0b, Wih0b, b0, F_);
    p0 ^= 1;
    lstm_step_kernel<<<256, 256, 0, stream>>>(h1[p1], h0[p0], c1,
                                              h1[p1 ^ 1], Whh1b, Wih1b, b1, H_);
    p1 ^= 1;
    proj_kernel<<<(B_ * F_) / 256, 256, 0, stream>>>(h1[p1], dec_w, dec_b, y, fut, s);
  }

  // ---- deconv -> d_out [B_,1,49] ----
  deconv_kernel<<<(B_ * (NSTEPS_ + KD_ - 1) + 255) / 256, 256, 0, stream>>>(
      fut, deconvw, (float*)d_out);
}
